// MHA_13125420057222
// MI455X (gfx1250) — compile-verified, matrix operand order checked by reading, not run on hardware
//
#include <hip/hip_runtime.h>
#include <hip/hip_bf16.h>
#include <stdint.h>

// MHA: B=2, S=2048, D=1024, H=16, DK=DV=64, DO=1024
#define Bb  2
#define Ss  2048
#define Dd  1024
#define Hh  16
#define DKk 64
#define DVv 64
#define DOo 1024

typedef __attribute__((ext_vector_type(16))) __bf16 v16bf;
typedef __attribute__((ext_vector_type(8)))  __bf16 v8bf;
typedef __attribute__((ext_vector_type(4)))  __bf16 v4bf;
typedef __attribute__((ext_vector_type(8)))  float  v8f;

__device__ inline __bf16 f2bf(float f) {
  union { float f; uint32_t u; } v; v.f = f;
  uint32_t r = v.u + 0x7FFFu + ((v.u >> 16) & 1u);   // round-to-nearest-even
  union { uint16_t u; __bf16 b; } o; o.u = (uint16_t)(r >> 16);
  return o.b;
}

// Async copy of one 16B chunk global->LDS (GV mode, ASYNCcnt-tracked).
// lds_off = LDS byte address (low 32 bits of generic LDS pointer).
__device__ inline void async_copy_b128(uint32_t lds_off, const void* gptr) {
  asm volatile("global_load_async_to_lds_b128 %0, %1, off"
               :: "v"(lds_off), "v"((uint64_t)(uintptr_t)gptr)
               : "memory");
}
__device__ inline void async_wait0() {
  asm volatile("s_wait_asynccnt 0x0" ::: "memory");
}

// Build a 16x32 bf16 WMMA fragment from LDS (rows of contiguous K, stride ldk).
// ISA 7.12.2 16-bit A layout: lane<16 -> row=lane, K = k0+[0..7] and k0+[16..23];
// lane>=16 -> row=lane-16, K = k0+[8..15] and k0+[24..31]. B fragments use the
// same shape from a [N][K] transposed-staged tile.
__device__ inline v16bf load_frag(const __bf16* lds, int ldk, int row0, int k0) {
  const int lane = threadIdx.x & 31;
  const __bf16* p = lds + (row0 + (lane & 15)) * ldk + k0 + ((lane >> 4) << 3);
  v8bf lo = *(const v8bf*)p;          // 16B ds_load
  v8bf hi = *(const v8bf*)(p + 16);   // 16B ds_load
  v16bf r;
#pragma unroll
  for (int i = 0; i < 8; ++i) { r[i] = lo[i]; r[i + 8] = hi[i]; }
  return r;
}

__device__ inline v8f wmma_bf16(v16bf a, v16bf b, v8f c) {
  return __builtin_amdgcn_wmma_f32_16x16x32_bf16(false, a, false, b, (short)0, c,
                                                 false, false);
}

__device__ inline v8f vzero8() { v8f z = {0.f,0.f,0.f,0.f,0.f,0.f,0.f,0.f}; return z; }

// ---------------------------------------------------------------------------
// K1: C_bf16[4096,1024] = bf16(A_f32[4096,1024]) @ bf16(W_f32[1024,1024])
// block tile 128x128, 8 waves (4x2), wave tile 32x64 (2x4 WMMA tiles), Kstep 32
// f32 sources are converted on the fly -> regular staged path (no async).
// ---------------------------------------------------------------------------
__global__ __launch_bounds__(256) void proj_kernel(const float* __restrict__ A,
                                                   const float* __restrict__ W,
                                                   __bf16* __restrict__ C) {
  __shared__ __attribute__((aligned(16))) __bf16 As[128 * 40];  // [row][k]
  __shared__ __attribute__((aligned(16))) __bf16 Bs[128 * 40];  // [n][k] transposed
  const int t = threadIdx.x;
  const int lane = t & 31, wid = t >> 5;
  const int wm = wid >> 1, wn = wid & 1;
  const int m0 = blockIdx.y * 128, n0 = blockIdx.x * 128;

  v8f acc[2][4];
#pragma unroll
  for (int mi = 0; mi < 2; ++mi)
#pragma unroll
    for (int ni = 0; ni < 4; ++ni) acc[mi][ni] = vzero8();

  for (int kt = 0; kt < Dd; kt += 32) {
    __syncthreads();
    // stage A: 128x32 f32 -> bf16 (coalesced float4 loads)
#pragma unroll
    for (int i = 0; i < 4; ++i) {
      int c = t + 256 * i;
      int row = c >> 3, col4 = (c & 7) * 4;
      float4 f = *(const float4*)(A + (size_t)(m0 + row) * Dd + kt + col4);
      v4bf b; b[0] = f2bf(f.x); b[1] = f2bf(f.y); b[2] = f2bf(f.z); b[3] = f2bf(f.w);
      *(v4bf*)&As[row * 40 + col4] = b;
    }
    // stage W transposed: rows kt..kt+32, cols n0..n0+128 -> Bs[n][k]
#pragma unroll
    for (int i = 0; i < 4; ++i) {
      int c = t + 256 * i;
      int kr = c >> 5, col4 = (c & 31) * 4;
      float4 f = *(const float4*)(W + (size_t)(kt + kr) * DOo + n0 + col4);
      Bs[(col4 + 0) * 40 + kr] = f2bf(f.x);
      Bs[(col4 + 1) * 40 + kr] = f2bf(f.y);
      Bs[(col4 + 2) * 40 + kr] = f2bf(f.z);
      Bs[(col4 + 3) * 40 + kr] = f2bf(f.w);
    }
    __syncthreads();
    v16bf a[2], b[4];
#pragma unroll
    for (int mi = 0; mi < 2; ++mi) a[mi] = load_frag(As, 40, wm * 32 + mi * 16, 0);
#pragma unroll
    for (int ni = 0; ni < 4; ++ni) b[ni] = load_frag(Bs, 40, wn * 64 + ni * 16, 0);
#pragma unroll
    for (int mi = 0; mi < 2; ++mi)
#pragma unroll
      for (int ni = 0; ni < 4; ++ni) acc[mi][ni] = wmma_bf16(a[mi], b[ni], acc[mi][ni]);
  }
  // epilogue: C/D layout -> lane<16: M=vgpr, lane>=16: M=8+vgpr; N=lane&15
#pragma unroll
  for (int mi = 0; mi < 2; ++mi)
#pragma unroll
    for (int ni = 0; ni < 4; ++ni) {
      int col = n0 + wn * 64 + ni * 16 + (lane & 15);
      int rb = m0 + wm * 32 + mi * 16 + ((lane >> 4) << 3);
#pragma unroll
      for (int r = 0; r < 8; ++r)
        C[(size_t)(rb + r) * DOo + col] = f2bf(acc[mi][ni][r]);
    }
}

// ---------------------------------------------------------------------------
// K2: scores[hb, i, j] = (qh[b,i,h,:] . kh[b,j,h,:]) / 8 + mask[b,i,j]  (f32)
// block tile 128x128, K=64 staged whole; hb = h*B + b = blockIdx.z
// q/k tiles are pure bf16 copies -> staged via GLOBAL_LOAD_ASYNC_TO_LDS_B128.
// ---------------------------------------------------------------------------
__global__ __launch_bounds__(256) void scores_kernel(const __bf16* __restrict__ qh,
                                                     const __bf16* __restrict__ kh,
                                                     const float* __restrict__ mask,
                                                     float* __restrict__ attn) {
  __shared__ __attribute__((aligned(16))) __bf16 Qs[128 * 72];
  __shared__ __attribute__((aligned(16))) __bf16 Ks[128 * 72];
  const int t = threadIdx.x;
  const int lane = t & 31, wid = t >> 5;
  const int wm = wid >> 1, wn = wid & 1;
  const int hb = blockIdx.z, h = hb >> 1, b = hb & 1;
  const int m0 = blockIdx.y * 128, n0 = blockIdx.x * 128;

  // stage qh / kh tiles (bf16, K=64 contiguous) via async global->LDS DMA
#pragma unroll
  for (int i = 0; i < 4; ++i) {
    int c = t + 256 * i;
    int row = c >> 3, col8 = (c & 7) * 8;
    const __bf16* gq = qh + (size_t)(b * Ss + m0 + row) * (Hh * DKk) + h * DKk + col8;
    const __bf16* gk = kh + (size_t)(b * Ss + n0 + row) * (Hh * DKk) + h * DKk + col8;
    async_copy_b128((uint32_t)(uintptr_t)&Qs[row * 72 + col8], gq);
    async_copy_b128((uint32_t)(uintptr_t)&Ks[row * 72 + col8], gk);
  }
  async_wait0();
  __syncthreads();

  v8f acc[2][4];
#pragma unroll
  for (int mi = 0; mi < 2; ++mi)
#pragma unroll
    for (int ni = 0; ni < 4; ++ni) acc[mi][ni] = vzero8();

#pragma unroll
  for (int k0 = 0; k0 < DKk; k0 += 32) {
    v16bf a[2], bfr[4];
#pragma unroll
    for (int mi = 0; mi < 2; ++mi) a[mi] = load_frag(Qs, 72, wm * 32 + mi * 16, k0);
#pragma unroll
    for (int ni = 0; ni < 4; ++ni) bfr[ni] = load_frag(Ks, 72, wn * 64 + ni * 16, k0);
#pragma unroll
    for (int mi = 0; mi < 2; ++mi)
#pragma unroll
      for (int ni = 0; ni < 4; ++ni) acc[mi][ni] = wmma_bf16(a[mi], bfr[ni], acc[mi][ni]);
  }

#pragma unroll
  for (int mi = 0; mi < 2; ++mi)
#pragma unroll
    for (int ni = 0; ni < 4; ++ni) {
      int col = n0 + wn * 64 + ni * 16 + (lane & 15);
      int rb = m0 + wm * 32 + mi * 16 + ((lane >> 4) << 3);
#pragma unroll
      for (int r = 0; r < 8; ++r) {
        float val = acc[mi][ni][r] * 0.125f +
                    mask[(size_t)b * Ss * Ss + (size_t)(rb + r) * Ss + col];
        attn[((size_t)hb * Ss + (rb + r)) * Ss + col] = val;
      }
    }
}

// ---------------------------------------------------------------------------
// K3: numerically stable row softmax in place; one block per row (2048 cols)
// ---------------------------------------------------------------------------
__global__ __launch_bounds__(256) void softmax_kernel(float* __restrict__ attn) {
  float* p = attn + (size_t)blockIdx.x * Ss;
  const int t = threadIdx.x;
  __shared__ float red[256];
  float v[8];
  float m = -1e30f;
#pragma unroll
  for (int i = 0; i < 8; ++i) { v[i] = p[t + 256 * i]; m = fmaxf(m, v[i]); }
  red[t] = m; __syncthreads();
  for (int s = 128; s > 0; s >>= 1) { if (t < s) red[t] = fmaxf(red[t], red[t + s]); __syncthreads(); }
  m = red[0]; __syncthreads();
  float sum = 0.f;
#pragma unroll
  for (int i = 0; i < 8; ++i) { v[i] = __expf(v[i] - m); sum += v[i]; }
  red[t] = sum; __syncthreads();
  for (int s = 128; s > 0; s >>= 1) { if (t < s) red[t] += red[t + s]; __syncthreads(); }
  float inv = 1.0f / red[0];
#pragma unroll
  for (int i = 0; i < 8; ++i) p[t + 256 * i] = v[i] * inv;
}

// ---------------------------------------------------------------------------
// K4: ctx[b,i,h,:] = attn[hb,i,:] @ vh[b,:,h,:]   (M=2048,N=64,K=2048 per hb)
// block tile 128x64, wave tile 32x32 (2x2), attn converted f32->bf16 on load
// gridDim = (Mtiles=16, hb=32)
// ---------------------------------------------------------------------------
__global__ __launch_bounds__(256) void attnv_kernel(const float* __restrict__ attn,
                                                    const __bf16* __restrict__ vh,
                                                    __bf16* __restrict__ ctx) {
  __shared__ __attribute__((aligned(16))) __bf16 As[128 * 40];  // attn tile [row][k]
  __shared__ __attribute__((aligned(16))) __bf16 Vs[64 * 40];   // v tile [n][k] transposed
  const int t = threadIdx.x;
  const int lane = t & 31, wid = t >> 5;
  const int wm = wid >> 1, wn = wid & 1;
  const int hb = blockIdx.y, h = hb >> 1, b = hb & 1;
  const int m0 = blockIdx.x * 128;
  const float* Ab = attn + (size_t)hb * Ss * Ss;

  v8f acc[2][2];
#pragma unroll
  for (int mi = 0; mi < 2; ++mi)
#pragma unroll
    for (int ni = 0; ni < 2; ++ni) acc[mi][ni] = vzero8();

  for (int kt = 0; kt < Ss; kt += 32) {
    __syncthreads();
#pragma unroll
    for (int i = 0; i < 4; ++i) {
      int c = t + 256 * i;
      int row = c >> 3, col4 = (c & 7) * 4;
      float4 f = *(const float4*)(Ab + (size_t)(m0 + row) * Ss + kt + col4);
      v4bf bb; bb[0] = f2bf(f.x); bb[1] = f2bf(f.y); bb[2] = f2bf(f.z); bb[3] = f2bf(f.w);
      *(v4bf*)&As[row * 40 + col4] = bb;
    }
    {  // vh rows kt..kt+32, cols h*64..+64 -> Vs[n][k]
      int k = t >> 3, col8 = (t & 7) * 8;
      v8bf d = *(const v8bf*)(vh + (size_t)(b * Ss + kt + k) * (Hh * DVv) + h * DVv + col8);
#pragma unroll
      for (int j = 0; j < 8; ++j) Vs[(col8 + j) * 40 + k] = d[j];
    }
    __syncthreads();
    v16bf a[2], bfr[2];
#pragma unroll
    for (int mi = 0; mi < 2; ++mi) a[mi] = load_frag(As, 40, wm * 32 + mi * 16, 0);
#pragma unroll
    for (int ni = 0; ni < 2; ++ni) bfr[ni] = load_frag(Vs, 40, wn * 32 + ni * 16, 0);
#pragma unroll
    for (int mi = 0; mi < 2; ++mi)
#pragma unroll
      for (int ni = 0; ni < 2; ++ni) acc[mi][ni] = wmma_bf16(a[mi], bfr[ni], acc[mi][ni]);
  }
#pragma unroll
  for (int mi = 0; mi < 2; ++mi)
#pragma unroll
    for (int ni = 0; ni < 2; ++ni) {
      int col = wn * 32 + ni * 16 + (lane & 15);
      int rb = m0 + wm * 32 + mi * 16 + ((lane >> 4) << 3);
#pragma unroll
      for (int r = 0; r < 8; ++r)
        ctx[(size_t)(b * Ss + rb + r) * (Hh * DVv) + h * DVv + col] = f2bf(acc[mi][ni][r]);
    }
}

// ---------------------------------------------------------------------------
// K5: out_f32[4096,1024] = ctx_bf16 @ bf16(Wo_f32) + bo
// ctx tile is a pure bf16 copy -> async global->LDS; Wo converts f32->bf16.
// ---------------------------------------------------------------------------
__global__ __launch_bounds__(256) void outproj_kernel(const __bf16* __restrict__ ctx,
                                                      const float* __restrict__ Wo,
                                                      const float* __restrict__ bo,
                                                      float* __restrict__ out) {
  __shared__ __attribute__((aligned(16))) __bf16 As[128 * 40];
  __shared__ __attribute__((aligned(16))) __bf16 Bs[128 * 40];
  const int t = threadIdx.x;
  const int lane = t & 31, wid = t >> 5;
  const int wm = wid >> 1, wn = wid & 1;
  const int m0 = blockIdx.y * 128, n0 = blockIdx.x * 128;

  v8f acc[2][4];
#pragma unroll
  for (int mi = 0; mi < 2; ++mi)
#pragma unroll
    for (int ni = 0; ni < 4; ++ni) acc[mi][ni] = vzero8();

  for (int kt = 0; kt < Hh * DVv; kt += 32) {
    __syncthreads();
#pragma unroll
    for (int i = 0; i < 2; ++i) {  // ctx bf16: 128x32 = 512 16B-chunks, async DMA
      int c = t + 256 * i;
      int row = c >> 2, col8 = (c & 3) * 8;
      const __bf16* g = ctx + (size_t)(m0 + row) * (Hh * DVv) + kt + col8;
      async_copy_b128((uint32_t)(uintptr_t)&As[row * 40 + col8], g);
    }
#pragma unroll
    for (int i = 0; i < 4; ++i) {  // Wo transposed, f32->bf16 on the fly
      int c = t + 256 * i;
      int kr = c >> 5, col4 = (c & 31) * 4;
      float4 f = *(const float4*)(Wo + (size_t)(kt + kr) * DOo + n0 + col4);
      Bs[(col4 + 0) * 40 + kr] = f2bf(f.x);
      Bs[(col4 + 1) * 40 + kr] = f2bf(f.y);
      Bs[(col4 + 2) * 40 + kr] = f2bf(f.z);
      Bs[(col4 + 3) * 40 + kr] = f2bf(f.w);
    }
    async_wait0();
    __syncthreads();
    v16bf a[2], bfr[4];
#pragma unroll
    for (int mi = 0; mi < 2; ++mi) a[mi] = load_frag(As, 40, wm * 32 + mi * 16, 0);
#pragma unroll
    for (int ni = 0; ni < 4; ++ni) bfr[ni] = load_frag(Bs, 40, wn * 64 + ni * 16, 0);
#pragma unroll
    for (int mi = 0; mi < 2; ++mi)
#pragma unroll
      for (int ni = 0; ni < 4; ++ni) acc[mi][ni] = wmma_bf16(a[mi], bfr[ni], acc[mi][ni]);
  }
#pragma unroll
  for (int mi = 0; mi < 2; ++mi)
#pragma unroll
    for (int ni = 0; ni < 4; ++ni) {
      int col = n0 + wn * 64 + ni * 16 + (lane & 15);
      int rb = m0 + wm * 32 + mi * 16 + ((lane >> 4) << 3);
      float bias = bo[col];
#pragma unroll
      for (int r = 0; r < 8; ++r)
        out[(size_t)(rb + r) * DOo + col] = acc[mi][ni][r] + bias;
    }
}

extern "C" void kernel_launch(void* const* d_in, const int* in_sizes, int n_in,
                              void* d_out, int out_size, void* d_ws, size_t ws_size,
                              hipStream_t stream) {
  (void)in_sizes; (void)n_in; (void)out_size; (void)ws_size;
  const float* q    = (const float*)d_in[0];
  const float* k    = (const float*)d_in[1];
  const float* v    = (const float*)d_in[2];
  const float* mask = (const float*)d_in[3];
  const float* Wq   = (const float*)d_in[4];
  const float* Wk   = (const float*)d_in[5];
  const float* Wv   = (const float*)d_in[6];
  const float* Wo   = (const float*)d_in[7];
  const float* bo   = (const float*)d_in[8];

  float* attn = (float*)d_out;                              // [H*B, S, S]
  float* out  = attn + (size_t)Hh * Bb * Ss * Ss;           // [B, S, DO]

  const size_t PROJ = (size_t)Bb * Ss * Hh * DKk;           // 4096*1024 elems
  __bf16* qh  = (__bf16*)d_ws;        // 8 MB each; total ws use = 32 MB
  __bf16* kh  = qh + PROJ;
  __bf16* vh  = kh + PROJ;
  __bf16* ctx = vh + PROJ;

  dim3 blk(256);
  proj_kernel<<<dim3(8, 32), blk, 0, stream>>>(q, Wq, qh);
  proj_kernel<<<dim3(8, 32), blk, 0, stream>>>(k, Wk, kh);
  proj_kernel<<<dim3(8, 32), blk, 0, stream>>>(v, Wv, vh);
  scores_kernel<<<dim3(16, 16, Hh * Bb), blk, 0, stream>>>(qh, kh, mask, attn);
  softmax_kernel<<<dim3(Hh * Bb * Ss), blk, 0, stream>>>(attn);
  attnv_kernel<<<dim3(16, Hh * Bb), blk, 0, stream>>>(attn, vh, ctx);
  outproj_kernel<<<dim3(8, 32), blk, 0, stream>>>(ctx, Wo, bo, out);
}
